// FNO_28561532519093
// MI455X (gfx1250) — matrix operand order repaired and checked by reading, not compiled, not run on hardware
//
#include <hip/hip_runtime.h>
#include <hip/hip_bf16.h>
#include <math.h>

// ---------------- types for WMMA (gfx1250, wave32) ----------------
typedef __bf16 bf16_t;
typedef bf16_t v16bf __attribute__((ext_vector_type(16)));
typedef bf16_t v8bf  __attribute__((ext_vector_type(8)));
typedef bf16_t v4bf  __attribute__((ext_vector_type(4)));
typedef float  v8f   __attribute__((ext_vector_type(8)));
typedef float  v4f   __attribute__((ext_vector_type(4)));

#define WMMA_BF16(A, B, C) \
  __builtin_amdgcn_wmma_f32_16x16x32_bf16(false, (A), false, (B), (short)0, (C), false, false)

// ---------------- problem dims ----------------
#define N_   16
#define NT_  128
#define NX_  128
#define H_   64
#define M_   12
#define DH_  128
#define DIN_ 128
#define KT2  24                 // 2*M kept t-modes (rows 0..11 and 116..127)
#define PI_  3.14159265358979323846f

// ---------------- workspace layout (float offsets) ----------------
static constexpr size_t OFF_TBX   = 0;                        // [32][128] fwd x-DFT basis (col-major: [col][x])
static constexpr size_t OFF_FTRE  = OFF_TBX   + 32 * 128;     // [24][128] fwd t-DFT re
static constexpr size_t OFF_FTIM  = OFF_FTRE  + 24 * 128;     // [24][128] fwd t-DFT im
static constexpr size_t OFF_IFTRE = OFF_FTIM  + 24 * 128;     // [128][24] inv t-DFT re
static constexpr size_t OFF_IFTIM = OFF_IFTRE + 128 * 24;     // [128][24] inv t-DFT im
static constexpr size_t OFF_TBIX  = OFF_IFTIM + 128 * 24;     // [128][32] inv x basis, transposed ([x][k])
static constexpr size_t OFF_H0    = OFF_TBIX  + 128 * 32;     // [16][64][128][128]
static constexpr size_t OFF_H1    = OFF_H0    + 16777216;
static constexpr size_t OFF_G1    = OFF_H1    + 16777216;     // [1024][128][24] (re/im interleave)
static constexpr size_t OFF_G2    = OFF_G1    + 3145728;      // [1024][24][12]*2
static constexpr size_t OFF_OFT   = OFF_G2    + 589824;       // [1024][24][12]*2
static constexpr size_t OFF_HT1   = OFF_OFT   + 589824;       // [1024][128][24]
static constexpr size_t OFF_X1    = OFF_HT1   + 3145728;      // [16][64][128][128]

// ---------------- WMMA fragment loaders (ISA §7.12.2 layouts) ----------------
// A 16x32 bf16, row-major LDS tile (lda multiple of 8 -> two 16B ds_load_b128).
// Lane l: m = l&15, g = l>>4; elements j<8 -> K=g*8+j, j>=8 -> K=16+g*8+(j-8).
__device__ __forceinline__ v16bf ldsA(const bf16_t* a, int lda) {
  const int l = threadIdx.x & 31;
  const int m = l & 15, g = l >> 4;
  const bf16_t* p = a + m * lda + g * 8;
  v8bf lo = *(const v8bf*)(p);        // K = g*8 .. g*8+7
  v8bf hi = *(const v8bf*)(p + 16);   // K = 16+g*8 .. 16+g*8+7
  return __builtin_shufflevector(lo, hi, 0, 1, 2, 3, 4, 5, 6, 7,
                                          8, 9, 10, 11, 12, 13, 14, 15);
}
// B 32x16 bf16 stored TRANSPOSED in LDS: Bt[n][k], contiguous in K.
// Lane l: n = l&15, g = l>>4; elements j<8 -> K=g*8+j, j>=8 -> K=16+g*8+(j-8).
__device__ __forceinline__ v16bf ldsBt(const bf16_t* bt, int ldb) {
  const int l = threadIdx.x & 31;
  const int n = l & 15, g = l >> 4;
  const bf16_t* p = bt + n * ldb + g * 8;
  v8bf lo = *(const v8bf*)(p);
  v8bf hi = *(const v8bf*)(p + 16);
  return __builtin_shufflevector(lo, hi, 0, 1, 2, 3, 4, 5, 6, 7,
                                          8, 9, 10, 11, 12, 13, 14, 15);
}

__device__ __forceinline__ float gelu_f(float v) {
  return 0.5f * v * (1.0f + erff(v * 0.70710678118654752f));
}

// vectorized f32 -> bf16x4 LDS fill (source and dest contiguous, 16B/8B aligned)
__device__ __forceinline__ void fill_bf16_v4(bf16_t* dst, const float* src, int nelem) {
  const v4f* s4 = (const v4f*)src;
  const int n4 = nelem >> 2;
  for (int e = threadIdx.x; e < n4; e += blockDim.x) {
    v4f s = s4[e];
    v4bf d = {(bf16_t)s.x, (bf16_t)s.y, (bf16_t)s.z, (bf16_t)s.w};
    *(v4bf*)(dst + e * 4) = d;
  }
}

// ---------------- DFT basis tables (built on device, deterministic) ----------------
__global__ void k_tables(float* ws) {
  float* TBX = ws + OFF_TBX;
  float* FTR = ws + OFF_FTRE;
  float* FTI = ws + OFF_FTIM;
  float* IFR = ws + OFF_IFTRE;
  float* IFI = ws + OFF_IFTIM;
  float* TBI = ws + OFF_TBIX;
  const float w0 = 2.0f * PI_ / 128.0f;
  // forward x basis, TRANSPOSED [col][x]: col 2*ky = cos, 2*ky+1 = -sin; cols 24..31 = 0
  for (int e = threadIdx.x; e < 32 * 128; e += blockDim.x) {
    int col = e >> 7, x = e & 127;
    float v = 0.0f;
    if (col < 24) {
      int ky = col >> 1;
      float th = w0 * (float)((ky * x) & 127);
      v = (col & 1) ? -sinf(th) : cosf(th);
    }
    TBX[e] = v;
  }
  // forward t basis (kt<12 -> freq kt, else freq 116+(kt-12)=104+kt)
  for (int e = threadIdx.x; e < 24 * 128; e += blockDim.x) {
    int kt = e >> 7, t = e & 127;
    int f = (kt < 12) ? kt : (104 + kt);
    float th = w0 * (float)((f * t) & 127);
    FTR[e] = cosf(th);
    FTI[e] = -sinf(th);
  }
  // inverse t basis e^{+i th} / Nt
  for (int e = threadIdx.x; e < 128 * 24; e += blockDim.x) {
    int t = e / 24, kt = e % 24;
    int f = (kt < 12) ? kt : (104 + kt);
    float th = w0 * (float)((f * t) & 127);
    IFR[e] = cosf(th) * (1.0f / 128.0f);
    IFI[e] = sinf(th) * (1.0f / 128.0f);
  }
  // inverse x basis, TRANSPOSED [x][k]: Re{(a+ib)e^{+ith}} weights; c_ky in {1,2}; /Nx
  for (int e = threadIdx.x; e < 128 * 32; e += blockDim.x) {
    int x = e >> 5, k = e & 31;
    float v = 0.0f;
    if (k < 24) {
      int ky = k >> 1;
      float c = (ky == 0) ? 1.0f : 2.0f;
      float th = w0 * (float)((ky * x) & 127);
      v = ((k & 1) ? -sinf(th) : cosf(th)) * (c / 128.0f);
    }
    TBI[e] = v;
  }
}

// ---------------- fc0: [x,t,u0] (K=130 pad 160) -> H=64, write [n][c][t][x] ----------------
__global__ void k_fc0(const float* __restrict__ t, const float* __restrict__ x,
                      const float* __restrict__ u0, const float* __restrict__ w,
                      const float* __restrict__ b, float* __restrict__ hout) {
  __shared__ __align__(16) bf16_t As[16][160];
  __shared__ __align__(16) bf16_t Bt[64][160];   // transposed: [c][k] == fc0_w layout
  const int p = blockIdx.x;       // 16384 tiles of 16 points
  const int n = p >> 10;
  const int rem = p & 1023;
  const int it = rem >> 3;
  const int x0 = (rem & 7) << 4;
  const int tid = threadIdx.x;
  for (int e = tid; e < 64 * 160; e += blockDim.x) {
    int c = e / 160, k = e % 160;
    Bt[c][k] = (bf16_t)((k < 130) ? w[c * 130 + k] : 0.0f);
  }
  const float tv = t[it];
  for (int e = tid; e < 16 * 160; e += blockDim.x) {
    int r = e / 160, k = e % 160;
    float v;
    if (k == 0)        v = x[x0 + r];
    else if (k == 1)   v = tv;
    else if (k < 130)  v = u0[n * DIN_ + (k - 2)];
    else               v = 0.0f;
    As[r][k] = (bf16_t)v;
  }
  __syncthreads();
  const int wv = tid >> 5;
  const int c0 = wv << 4;
  v8f acc = {};
#pragma unroll
  for (int ks = 0; ks < 5; ++ks) {
    v16bf af = ldsA(&As[0][ks * 32], 160);
    v16bf bf = ldsBt(&Bt[c0][ks * 32], 160);
    acc = WMMA_BF16(af, bf, acc);
  }
  const int l = tid & 31, nn = l & 15, g = l >> 4;
  const int c = c0 + nn;
  const float bias = b[c];
#pragma unroll
  for (int r = 0; r < 8; ++r) {
    int m = r + 8 * g;            // x offset in tile
    hout[(((size_t)(n * H_ + c) * NT_ + it) * NX_) + x0 + m] = acc[r] + bias;
  }
}

// ---------------- forward x-DFT per image: [128t x 128x] @ [128x x 24] ----------------
__global__ void k_dftx(const float* __restrict__ hin, const float* __restrict__ TBX,
                       float* __restrict__ G1) {
  __shared__ __align__(16) bf16_t As[128][128];
  __shared__ __align__(16) bf16_t Bt[32][128];   // transposed: [col][x] == TBX layout
  const int img = blockIdx.x;     // n*64+c
  const float* src = hin + (size_t)img * (NT_ * NX_);
  fill_bf16_v4(&As[0][0], src, 128 * 128);
  fill_bf16_v4(&Bt[0][0], TBX, 32 * 128);
  __syncthreads();
  const int wv = threadIdx.x >> 5;          // 8 waves -> 8 M-tiles
  const int l = threadIdx.x & 31, nn = l & 15, g = l >> 4;
#pragma unroll
  for (int nt = 0; nt < 2; ++nt) {
    v8f acc = {};
#pragma unroll
    for (int ks = 0; ks < 4; ++ks) {
      v16bf af = ldsA(&As[wv * 16][ks * 32], 128);
      v16bf bf = ldsBt(&Bt[nt * 16][ks * 32], 128);
      acc = WMMA_BF16(af, bf, acc);
    }
    const int col = nt * 16 + nn;
    if (col < 24) {
#pragma unroll
      for (int r = 0; r < 8; ++r) {
        int trow = wv * 16 + r + 8 * g;
        G1[((size_t)img * NT_ + trow) * KT2 + col] = acc[r];
      }
    }
  }
}

// ---------------- forward t-DFT on 24 kept rows (tiny, VALU) ----------------
__global__ void k_dftt(const float* __restrict__ G1, const float* __restrict__ FTR,
                       const float* __restrict__ FTI, float* __restrict__ G2) {
  int tid = blockIdx.x * blockDim.x + threadIdx.x;   // 294912 exact
  int ky = tid % 12;
  int kt = (tid / 12) % 24;
  int nc = tid / 288;
  float ar = 0.f, ai = 0.f;
  const float* g = G1 + (size_t)nc * NT_ * KT2;
  for (int t = 0; t < NT_; ++t) {
    float gr = g[t * KT2 + 2 * ky], gi = g[t * KT2 + 2 * ky + 1];
    float fr = FTR[kt * NT_ + t], fi = FTI[kt * NT_ + t];
    ar += fr * gr - fi * gi;
    ai += fr * gi + fi * gr;
  }
  size_t o = ((size_t)(nc * 24 + kt) * 12 + ky) * 2;
  G2[o] = ar;
  G2[o + 1] = ai;
}

// ---------------- per-mode complex channel mix: [16n x 64i] @ [64i x 64o] ----------------
__global__ void k_specmul(const float* __restrict__ G2, const float* __restrict__ w1r,
                          const float* __restrict__ w1i, const float* __restrict__ w2r,
                          const float* __restrict__ w2i, int layer, float* __restrict__ OFT) {
  __shared__ __align__(16) bf16_t Ar[16][64], Ai[16][64];
  __shared__ __align__(16) bf16_t Brt[64][64], Bit[64][64], Bnt[64][64]; // transposed [o][i]
  const int mode = blockIdx.x;            // 288 modes
  const int kt = mode / 12, ky = mode % 12;
  const int reg = (kt < 12) ? 0 : 1;
  const int kx = reg ? (kt - 12) : kt;
  const float* wr = reg ? w2r : w1r;
  const float* wi = reg ? w2i : w1i;
  for (int e = threadIdx.x; e < 16 * 64; e += blockDim.x) {
    int n = e >> 6, i = e & 63;
    size_t base = ((size_t)((n * 64 + i) * 24 + kt) * 12 + ky) * 2;
    Ar[n][i] = (bf16_t)G2[base];
    Ai[n][i] = (bf16_t)G2[base + 1];
  }
  for (int e = threadIdx.x; e < 64 * 64; e += blockDim.x) {
    int o = e >> 6, i = e & 63;
    size_t widx = (size_t)((layer * 64 + i) * 64 + o) * 144 + kx * 12 + ky;
    float re = wr[widx], im = wi[widx];
    Brt[o][i] = (bf16_t)re;
    Bit[o][i] = (bf16_t)im;
    Bnt[o][i] = (bf16_t)(-im);            // bf16 WMMA has no A/B neg bits -> negated copy
  }
  __syncthreads();
  const int wv = threadIdx.x >> 5;
  const int o0 = wv << 4;
  v8f accR = {}, accI = {};
#pragma unroll
  for (int ks = 0; ks < 2; ++ks) {
    v16bf aR = ldsA(&Ar[0][ks * 32], 64);
    v16bf aI = ldsA(&Ai[0][ks * 32], 64);
    v16bf bR = ldsBt(&Brt[o0][ks * 32], 64);
    v16bf bI = ldsBt(&Bit[o0][ks * 32], 64);
    v16bf bN = ldsBt(&Bnt[o0][ks * 32], 64);
    accR = WMMA_BF16(aR, bR, accR);
    accR = WMMA_BF16(aI, bN, accR);
    accI = WMMA_BF16(aR, bI, accI);
    accI = WMMA_BF16(aI, bR, accI);
  }
  const int l = threadIdx.x & 31, nn = l & 15, g = l >> 4;
  const int o = o0 + nn;
#pragma unroll
  for (int r = 0; r < 8; ++r) {
    int n = r + 8 * g;
    size_t ob = ((size_t)((n * 64 + o) * 24 + kt) * 12 + ky) * 2;
    OFT[ob] = accR[r];
    OFT[ob + 1] = accI[r];
  }
}

// ---------------- inverse t-DFT (tiny, VALU) ----------------
__global__ void k_idft_t(const float* __restrict__ OFT, const float* __restrict__ IFR,
                         const float* __restrict__ IFI, float* __restrict__ HT1) {
  int tid = blockIdx.x * blockDim.x + threadIdx.x;   // 1572864 exact
  int ky = tid % 12;
  int t = (tid / 12) % 128;
  int nc = tid / 1536;
  float ar = 0.f, ai = 0.f;
  for (int kt = 0; kt < 24; ++kt) {
    size_t ob = ((size_t)(nc * 24 + kt) * 12 + ky) * 2;
    float fr = IFR[t * 24 + kt], fi = IFI[t * 24 + kt];
    float vr = OFT[ob], vi = OFT[ob + 1];
    ar += fr * vr - fi * vi;
    ai += fr * vi + fi * vr;
  }
  size_t hb = ((size_t)nc * 128 + t) * KT2;
  HT1[hb + 2 * ky] = ar;
  HT1[hb + 2 * ky + 1] = ai;
}

// ---------------- inverse x reconstruction: [128t x 24] @ [24 x 128x] ----------------
__global__ void k_idftx(const float* __restrict__ HT1, const float* __restrict__ TBI,
                        float* __restrict__ X1) {
  __shared__ __align__(16) bf16_t As[128][32];
  __shared__ __align__(16) bf16_t Bt[128][32];   // transposed: [x][k] == TBI layout
  const int img = blockIdx.x;
  const float* src = HT1 + (size_t)img * NT_ * KT2;
  for (int e = threadIdx.x; e < 128 * 32; e += blockDim.x) {
    int t = e >> 5, k = e & 31;
    (&As[0][0])[e] = (bf16_t)((k < 24) ? src[t * KT2 + k] : 0.0f);
  }
  fill_bf16_v4(&Bt[0][0], TBI, 128 * 32);
  __syncthreads();
  const int wv = threadIdx.x >> 5;
  const int l = threadIdx.x & 31, nn = l & 15, g = l >> 4;
  v16bf af = ldsA(&As[wv * 16][0], 32);
  float* dst = X1 + (size_t)img * NT_ * NX_;
#pragma unroll
  for (int nt = 0; nt < 8; ++nt) {
    v8f acc = {};
    v16bf bf = ldsBt(&Bt[nt * 16][0], 32);
    acc = WMMA_BF16(af, bf, acc);
#pragma unroll
    for (int r = 0; r < 8; ++r) {
      int trow = wv * 16 + r + 8 * g;
      dst[trow * NX_ + nt * 16 + nn] = acc[r];
    }
  }
}

// ---------------- pointwise conv [64o x 64c] @ [64c x 128s] + X1 + bias (+gelu) ----------------
__global__ void k_conv(const float* __restrict__ hin, const float* __restrict__ X1,
                       const float* __restrict__ wc, const float* __restrict__ bc,
                       int layer, int do_gelu, float* __restrict__ hout) {
  __shared__ __align__(16) bf16_t Aw[64][64];
  __shared__ __align__(16) bf16_t Bht[128][64];   // transposed: [s][c]
  const int n = blockIdx.x >> 7;
  const int s0 = (blockIdx.x & 127) << 7;
  // prefetch the X1 lines consumed in the epilogue (gfx1250 global_prefetch)
  __builtin_prefetch(&X1[(((size_t)n * 64 + (threadIdx.x >> 2)) << 14) + s0], 0, 0);
  for (int e = threadIdx.x; e < 64 * 64; e += blockDim.x) {
    int o = e >> 6, c = e & 63;
    Aw[o][c] = (bf16_t)wc[(size_t)(layer * 64 + o) * 64 + c];
  }
  for (int e = threadIdx.x; e < 64 * 128; e += blockDim.x) {
    int c = e >> 7, j = e & 127;   // contiguous global reads along j
    Bht[j][c] = (bf16_t)hin[(((size_t)n * 64 + c) << 14) + s0 + j];
  }
  __syncthreads();
  const int wv = threadIdx.x >> 5;
  const int mt = wv & 3;
  const int nh = wv >> 2;
  const int l = threadIdx.x & 31, nn = l & 15, g = l >> 4;
  v16bf af0 = ldsA(&Aw[mt * 16][0], 64);
  v16bf af1 = ldsA(&Aw[mt * 16][32], 64);
#pragma unroll
  for (int jn = 0; jn < 4; ++jn) {
    const int nt = nh * 4 + jn;
    v8f acc = {};
    acc = WMMA_BF16(af0, ldsBt(&Bht[nt * 16][0], 64), acc);
    acc = WMMA_BF16(af1, ldsBt(&Bht[nt * 16][32], 64), acc);
#pragma unroll
    for (int r = 0; r < 8; ++r) {
      const int o = mt * 16 + r + 8 * g;
      const int s = s0 + nt * 16 + nn;
      size_t gi = (((size_t)n * 64 + o) << 14) + s;
      float v = acc[r] + X1[gi] + bc[layer * 64 + o];
      if (do_gelu) v = gelu_f(v);
      hout[gi] = v;
    }
  }
}

// ---------------- head: gelu(h @ fc1^T + b1) @ fc2^T + b2 ----------------
__global__ void k_head(const float* __restrict__ hin, const float* __restrict__ w1,
                       const float* __restrict__ b1, const float* __restrict__ w2,
                       const float* __restrict__ b2, float* __restrict__ out) {
  __shared__ __align__(16) bf16_t Ah[16][64];
  __shared__ __align__(16) bf16_t Bwt[128][64];   // transposed: [d][c] == fc1_w layout
  __shared__ float Z[16][128];
  const int p0 = blockIdx.x << 4;
  const int n = p0 >> 14;
  const int s = p0 & 16383;
  for (int e = threadIdx.x; e < 16 * 64; e += blockDim.x) {
    int r = e >> 6, c = e & 63;
    Ah[r][c] = (bf16_t)hin[(((size_t)n * 64 + c) << 14) + s + r];
  }
  fill_bf16_v4(&Bwt[0][0], w1, 128 * 64);
  __syncthreads();
  const int wv = threadIdx.x >> 5;
  const int l = threadIdx.x & 31, nn = l & 15, g = l >> 4;
#pragma unroll
  for (int jn = 0; jn < 2; ++jn) {
    const int nt = wv * 2 + jn;
    v8f acc = {};
    acc = WMMA_BF16(ldsA(&Ah[0][0], 64),  ldsBt(&Bwt[nt * 16][0], 64), acc);
    acc = WMMA_BF16(ldsA(&Ah[0][32], 64), ldsBt(&Bwt[nt * 16][32], 64), acc);
    const int d = nt * 16 + nn;
    const float bias = b1[d];
#pragma unroll
    for (int r = 0; r < 8; ++r) Z[r + 8 * g][d] = gelu_f(acc[r] + bias);
  }
  __syncthreads();
  if (threadIdx.x < 16) {
    float sum = 0.f;
    for (int d = 0; d < 128; ++d) sum += w2[d] * Z[threadIdx.x][d];
    out[(size_t)n * 16384 + s + threadIdx.x] = sum + b2[0];
  }
}

// ---------------- launch ----------------
extern "C" void kernel_launch(void* const* d_in, const int* in_sizes, int n_in,
                              void* d_out, int out_size, void* d_ws, size_t ws_size,
                              hipStream_t stream) {
  (void)in_sizes; (void)n_in; (void)out_size; (void)ws_size;
  const float* t   = (const float*)d_in[0];
  const float* x   = (const float*)d_in[1];
  const float* u0  = (const float*)d_in[2];
  const float* f0w = (const float*)d_in[3];
  const float* f0b = (const float*)d_in[4];
  const float* w1r = (const float*)d_in[5];
  const float* w1i = (const float*)d_in[6];
  const float* w2r = (const float*)d_in[7];
  const float* w2i = (const float*)d_in[8];
  const float* wc  = (const float*)d_in[9];
  const float* bc  = (const float*)d_in[10];
  const float* f1w = (const float*)d_in[11];
  const float* f1b = (const float*)d_in[12];
  const float* f2w = (const float*)d_in[13];
  const float* f2b = (const float*)d_in[14];
  float* ws = (float*)d_ws;

  k_tables<<<1, 256, 0, stream>>>(ws);
  k_fc0<<<16384, 128, 0, stream>>>(t, x, u0, f0w, f0b, ws + OFF_H0);

  float* cur = ws + OFF_H0;
  float* nxt = ws + OFF_H1;
  for (int l = 0; l < 4; ++l) {
    k_dftx<<<1024, 256, 0, stream>>>(cur, ws + OFF_TBX, ws + OFF_G1);
    k_dftt<<<1152, 256, 0, stream>>>(ws + OFF_G1, ws + OFF_FTRE, ws + OFF_FTIM, ws + OFF_G2);
    k_specmul<<<288, 128, 0, stream>>>(ws + OFF_G2, w1r, w1i, w2r, w2i, l, ws + OFF_OFT);
    k_idft_t<<<6144, 256, 0, stream>>>(ws + OFF_OFT, ws + OFF_IFTRE, ws + OFF_IFTIM, ws + OFF_HT1);
    k_idftx<<<1024, 256, 0, stream>>>(ws + OFF_HT1, ws + OFF_TBIX, ws + OFF_X1);
    k_conv<<<2048, 256, 0, stream>>>(cur, ws + OFF_X1, wc, bc, l, (l < 3) ? 1 : 0, nxt);
    float* tmp = cur; cur = nxt; nxt = tmp;
  }
  k_head<<<16384, 128, 0, stream>>>(cur, f1w, f1b, f2w, f2b, (float*)d_out);
}
